// MessageAggregationAttention_76398878261526
// MI455X (gfx1250) — matrix operand order repaired and checked by reading, not compiled, not run on hardware
//
#include <hip/hip_runtime.h>
#include <cstdint>

// ---------------- problem constants ----------------
#define B_  256
#define E_  16384
#define M_  65536
#define H_  128
#define NH_ 4
#define HD_ 32
#define LQ_ 128
#define LK_ 384

typedef _Float16 h16;
typedef _Float16 v16h __attribute__((ext_vector_type(16)));
typedef float    v8f  __attribute__((ext_vector_type(8)));
typedef float    v4f  __attribute__((ext_vector_type(4)));

union Frag { v16h v; v4f f4[2]; h16 h[16]; };

// ---------------- setup kernels ----------------

__global__ void searchsorted_kernel(const int* __restrict__ arr, int n, int* __restrict__ start) {
  int b = blockIdx.x * blockDim.x + threadIdx.x;
  if (b >= B_) return;
  int lo = 0, hi = n;
  while (lo < hi) { int mid = (lo + hi) >> 1; if (arr[mid] < b) lo = mid + 1; else hi = mid; }
  start[b] = lo;
}

__global__ void cvt_f16_kernel(const float* __restrict__ src, h16* __restrict__ dst, int n) {
  int i = blockIdx.x * blockDim.x + threadIdx.x;
  if (i < n) dst[i] = (h16)src[i];
}

__global__ void scatter_q_kernel(const float* __restrict__ edge_attr,
                                 const int* __restrict__ edge_batch,
                                 const int* __restrict__ startq,
                                 h16* __restrict__ qc, int* __restrict__ rowidx) {
  int e = blockIdx.x, t = threadIdx.x;
  int b = edge_batch[e];
  int p = e - startq[b]; if (p > LQ_ - 1) p = LQ_ - 1;
  qc[(long)(b * LQ_ + p) * H_ + t] = (h16)edge_attr[(long)e * H_ + t];
  if (t == 0) rowidx[e] = b * LQ_ + p;
}

__global__ void scatter_k_kernel(const float* __restrict__ edge_attr,
                                 const int* __restrict__ iel,
                                 const int* __restrict__ ieb,
                                 const int* __restrict__ startk,
                                 h16* __restrict__ kc, unsigned char* __restrict__ kmask) {
  int m = blockIdx.x, t = threadIdx.x;
  int b = ieb[m];
  int p = m - startk[b]; if (p > LK_ - 1) p = LK_ - 1;
  int src = iel[m];
  kc[(long)(b * LK_ + p) * H_ + t] = (h16)edge_attr[(long)src * H_ + t];
  if (t == 0) kmask[b * LK_ + p] = 1;
}

// ---------------- generic WMMA GEMM (32x32 tile per wave, 2x2 WMMA) ----------------
// out = op( A[M x K] @ W^T + bias ); B-fragment element (k,n) = W[n*ldw + k].
template<bool GATHER, bool RELU, bool HAS_RES, bool OUT16, bool OUT32>
__global__ __launch_bounds__(256)
void gemm_wmma_kernel(const h16* __restrict__ A, const int* __restrict__ rowidx, int lda,
                      const h16* __restrict__ W, int ldw,
                      const float* __restrict__ bias,
                      const float* __restrict__ res,
                      h16* __restrict__ outh, int ob_rows, long ob_stride, int o_rs, int o_cs,
                      float* __restrict__ outf,
                      int Mrows, int Ncols, int Kdim) {
  int lane = threadIdx.x & 31;
  int wv   = threadIdx.x >> 5;
  int tilesN = Ncols >> 5;
  int gid = blockIdx.x * (blockDim.x >> 5) + wv;
  int totalTiles = (Mrows >> 5) * tilesN;
  if (gid >= totalTiles) return;
  int tn = gid % tilesN, tm = gid / tilesN;
  int hs = lane >> 4;
  int ln = lane & 15;

  int ar0 = tm * 32 + ln, ar1 = ar0 + 16;
  long arow0 = GATHER ? (long)rowidx[ar0] : (long)ar0;
  long arow1 = GATHER ? (long)rowidx[ar1] : (long)ar1;
  const h16* ap0 = A + arow0 * (long)lda;
  const h16* ap1 = A + arow1 * (long)lda;
  const h16* bp0 = W + (long)(tn * 32 + ln) * ldw + hs * 16;  // contiguous 32B per lane
  const h16* bp1 = bp0 + (long)16 * ldw;
  int col0 = tn * 32 + ln, col1 = col0 + 16;

  float bi0 = bias[col0], bi1 = bias[col1];
  v8f c00, c01, c10, c11;
#pragma unroll
  for (int r = 0; r < 8; r++) { c00[r] = bi0; c01[r] = bi1; c10[r] = bi0; c11[r] = bi1; }

  for (int kb = 0; kb < Kdim; kb += 32) {
    Frag a0, a1, b0, b1;
    a0.f4[0] = *(const v4f*)(ap0 + kb + hs * 16);
    a0.f4[1] = *(const v4f*)(ap0 + kb + hs * 16 + 8);
    a1.f4[0] = *(const v4f*)(ap1 + kb + hs * 16);
    a1.f4[1] = *(const v4f*)(ap1 + kb + hs * 16 + 8);
    b0.f4[0] = *(const v4f*)(bp0 + kb);
    b0.f4[1] = *(const v4f*)(bp0 + kb + 8);
    b1.f4[0] = *(const v4f*)(bp1 + kb);
    b1.f4[1] = *(const v4f*)(bp1 + kb + 8);
    c00 = __builtin_amdgcn_wmma_f32_16x16x32_f16(false, a0.v, false, b0.v, (short)0, c00, false, false);
    c01 = __builtin_amdgcn_wmma_f32_16x16x32_f16(false, a0.v, false, b1.v, (short)0, c01, false, false);
    c10 = __builtin_amdgcn_wmma_f32_16x16x32_f16(false, a1.v, false, b0.v, (short)0, c10, false, false);
    c11 = __builtin_amdgcn_wmma_f32_16x16x32_f16(false, a1.v, false, b1.v, (short)0, c11, false, false);
  }

  auto emit = [&](v8f& cc, int rbase, int col) {
#pragma unroll
    for (int r = 0; r < 8; r++) {
      int grow = rbase + (hs ? 8 + r : r);
      float v = cc[r];
      if (RELU)    v = v > 0.f ? v : 0.f;
      if (HAS_RES) v += res[(long)grow * Ncols + col];
      if (OUT32)   outf[(long)grow * Ncols + col] = v;
      if (OUT16) {
        long oaddr = (long)(grow / ob_rows) * ob_stride + (long)(grow % ob_rows) * o_rs + (long)col * o_cs;
        outh[oaddr] = (h16)v;
      }
    }
  };
  emit(c00, tm * 32,      col0);
  emit(c01, tm * 32,      col1);
  emit(c10, tm * 32 + 16, col0);
  emit(c11, tm * 32 + 16, col1);
}

// ---------------- flash attention ----------------
// One wave per (batch, head, 16-row Q strip). Scale folded into Q fragment.
// Softmax stats computed in A-layout (lane L owns row L&15; partner half in lane L^16),
// so row reductions are in-register + one shfl_xor(16). Only the o-rescale needs
// per-row broadcasts (8 shfl per chunk).
__global__ __launch_bounds__(256)
void attention_kernel(const h16* __restrict__ Q, const h16* __restrict__ K,
                      const h16* __restrict__ Vt, const unsigned char* __restrict__ kmask,
                      h16* __restrict__ ctx) {
  __shared__ __align__(16) float sbuf[8][16 * 32];  // per-wave masked-S staging (f32)
  int lane = threadIdx.x & 31;
  int wv   = threadIdx.x >> 5;
  int gid  = blockIdx.x * 8 + wv;                   // B_*NH_*(LQ_/16) = 8192 waves
  int qt = gid & 7;
  int hh = (gid >> 3) & (NH_ - 1);
  int b  = gid >> 5;
  int hs = lane >> 4, ln = lane & 15;
  const float scale = 0.17677669529663687f;         // 1/sqrt(HD_)
  float* sb = sbuf[wv];

  // Q A-fragment with folded softmax scale
  Frag aq;
  {
    Frag t;
    const h16* qp = Q + (long)(b * LQ_ + qt * 16 + ln) * H_ + hh * HD_;
    t.f4[0] = *(const v4f*)(qp + hs * 16);
    t.f4[1] = *(const v4f*)(qp + hs * 16 + 8);
#pragma unroll
    for (int i = 0; i < 16; i++) aq.h[i] = (h16)((float)t.h[i] * scale);
  }

  v8f o0, o1;
#pragma unroll
  for (int r = 0; r < 8; r++) { o0[r] = 0.f; o1[r] = 0.f; }
  float m_ln = -__builtin_inff();   // running max of row (lane&15)
  float l_ln = 0.f;                 // running sum of row (lane&15)

  for (int ct = 0; ct < LK_ / 32; ct++) {
    int kbase = ct * 32;
    // K B-fragments: element (k,n) = K[b][kbase+n][hh*HD_+k] -> contiguous per lane
    Frag bk0, bk1;
    const h16* kp0 = K + (long)(b * LK_ + kbase + ln) * H_ + hh * HD_ + hs * 16;
    const h16* kp1 = kp0 + 16 * H_;
    bk0.f4[0] = *(const v4f*)kp0; bk0.f4[1] = *(const v4f*)(kp0 + 8);
    bk1.f4[0] = *(const v4f*)kp1; bk1.f4[1] = *(const v4f*)(kp1 + 8);

    v8f z;
#pragma unroll
    for (int r = 0; r < 8; r++) z[r] = 0.f;
    v8f s0 = __builtin_amdgcn_wmma_f32_16x16x32_f16(false, aq.v, false, bk0.v, (short)0, z, false, false);
    v8f s1 = __builtin_amdgcn_wmma_f32_16x16x32_f16(false, aq.v, false, bk1.v, (short)0, z, false, false);

    bool mk0 = kmask[b * LK_ + kbase + ln] != 0;
    bool mk1 = kmask[b * LK_ + kbase + ln + 16] != 0;

    // masked S (C-layout) -> LDS f32
#pragma unroll
    for (int r = 0; r < 8; r++) {
      int row = hs ? 8 + r : r;
      sb[row * 32 + ln]      = mk0 ? s0[r] : -1e9f;
      sb[row * 32 + ln + 16] = mk1 ? s1[r] : -1e9f;
    }

    // read my row's 16 values in A-layout order (partner half holds the other 16)
    const float* rp = sb + ln * 32 + hs * 16;
    v4f x0 = *(const v4f*)(rp);
    v4f x1 = *(const v4f*)(rp + 4);
    v4f x2 = *(const v4f*)(rp + 8);
    v4f x3 = *(const v4f*)(rp + 12);

    float cm = fmaxf(fmaxf(fmaxf(x0[0], x0[1]), fmaxf(x0[2], x0[3])),
                     fmaxf(fmaxf(x1[0], x1[1]), fmaxf(x1[2], x1[3])));
    cm = fmaxf(cm, fmaxf(fmaxf(fmaxf(x2[0], x2[1]), fmaxf(x2[2], x2[3])),
                         fmaxf(fmaxf(x3[0], x3[1]), fmaxf(x3[2], x3[3]))));
    cm = fmaxf(cm, __shfl_xor(cm, 16, 32));        // combine with partner half

    float mn  = fmaxf(m_ln, cm);
    float fac = __expf(m_ln - mn);
    m_ln = mn;

    Frag ap;
    float rs = 0.f;
#pragma unroll
    for (int i = 0; i < 4; i++) { float p = __expf(x0[i] - mn); ap.h[i]      = (h16)p; rs += p; }
#pragma unroll
    for (int i = 0; i < 4; i++) { float p = __expf(x1[i] - mn); ap.h[4 + i]  = (h16)p; rs += p; }
#pragma unroll
    for (int i = 0; i < 4; i++) { float p = __expf(x2[i] - mn); ap.h[8 + i]  = (h16)p; rs += p; }
#pragma unroll
    for (int i = 0; i < 4; i++) { float p = __expf(x3[i] - mn); ap.h[12 + i] = (h16)p; rs += p; }
    rs += __shfl_xor(rs, 16, 32);
    l_ln = l_ln * fac + rs;

    // rescale accumulators by per-row factor (broadcast from owning lane)
#pragma unroll
    for (int r = 0; r < 8; r++) {
      float fr = __shfl(fac, hs ? 8 + r : r, 32);
      o0[r] *= fr; o1[r] *= fr;
    }

    // V B-fragments from transposed V: element (k,n) = Vt[b][hh*HD_+n][kbase+k] -> contiguous
    Frag bv0, bv1;
    const h16* vp0 = Vt + (long)b * H_ * LK_ + (long)(hh * HD_ + ln) * LK_ + kbase + hs * 16;
    const h16* vp1 = vp0 + 16 * LK_;
    bv0.f4[0] = *(const v4f*)vp0; bv0.f4[1] = *(const v4f*)(vp0 + 8);
    bv1.f4[0] = *(const v4f*)vp1; bv1.f4[1] = *(const v4f*)(vp1 + 8);

    o0 = __builtin_amdgcn_wmma_f32_16x16x32_f16(false, ap.v, false, bv0.v, (short)0, o0, false, false);
    o1 = __builtin_amdgcn_wmma_f32_16x16x32_f16(false, ap.v, false, bv1.v, (short)0, o1, false, false);
  }

  float inv_l = 1.0f / l_ln;
#pragma unroll
  for (int r = 0; r < 8; r++) {
    float ir = __shfl(inv_l, hs ? 8 + r : r, 32);
    int row = qt * 16 + (hs ? 8 + r : r);
    h16* cp = ctx + (long)(b * LQ_ + row) * H_ + hh * HD_ + ln;
    cp[0]  = (h16)(o0[r] * ir);
    cp[16] = (h16)(o1[r] * ir);
  }
}

// ---------------- host launcher ----------------
static inline size_t al256(size_t x) { return (x + 255) & ~(size_t)255; }

extern "C" void kernel_launch(void* const* d_in, const int* in_sizes, int n_in,
                              void* d_out, int out_size, void* d_ws, size_t ws_size,
                              hipStream_t stream) {
  (void)in_sizes; (void)n_in; (void)out_size; (void)ws_size;
  const float* edge_attr = (const float*)d_in[1];
  const int*   iel       = (const int*)d_in[2];
  const int*   ieb       = (const int*)d_in[3];
  const int*   edge_batch= (const int*)d_in[4];
  const float* in_proj_w = (const float*)d_in[5];
  const float* in_proj_b = (const float*)d_in[6];
  const float* out_proj_w= (const float*)d_in[7];
  const float* out_proj_b= (const float*)d_in[8];
  const float* w1        = (const float*)d_in[9];
  const float* b1        = (const float*)d_in[10];
  const float* w2        = (const float*)d_in[11];
  const float* b2        = (const float*)d_in[12];

  char* ws = (char*)d_ws;
  size_t off = 0;
  auto carve = [&](size_t bytes) { void* p = ws + off; off = al256(off + bytes); return p; };

  h16* w16   = (h16*)carve((size_t)8 * H_ * H_ * 2);  // wq|wk|wv|wo|w1|w2 (f16)
  int* startq= (int*)carve((size_t)B_ * 4);
  int* startk= (int*)carve((size_t)B_ * 4);
  int* rowidx= (int*)carve((size_t)E_ * 4);
  unsigned char* kmask = (unsigned char*)carve((size_t)B_ * LK_);
  h16* qc = (h16*)carve((size_t)B_ * LQ_ * H_ * 2);   // dense q (f16); reused as ctx
  h16* kc = (h16*)carve((size_t)B_ * LK_ * H_ * 2);   // dense k (f16); reused for att/ffn bufs
  h16* Qb = (h16*)carve((size_t)B_ * LQ_ * H_ * 2);
  h16* Kb = (h16*)carve((size_t)B_ * LK_ * H_ * 2);
  h16* Vt = (h16*)carve((size_t)B_ * H_ * LK_ * 2);   // V transposed [B][H][LK]

  h16*   ctx   = qc;                                   // qc dead after Q projection
  float* att32 = (float*)kc;                           // kc dead after K/V projections
  h16*   att16 = (h16*)((char*)kc + al256((size_t)E_ * H_ * 4));
  h16*   h1b   = (h16*)((char*)att16 + al256((size_t)E_ * H_ * 2));

  hipMemsetAsync(qc, 0, (size_t)B_ * LQ_ * H_ * 2, stream);
  hipMemsetAsync(kc, 0, (size_t)B_ * LK_ * H_ * 2, stream);
  hipMemsetAsync(kmask, 0, (size_t)B_ * LK_, stream);

  cvt_f16_kernel<<<(3 * H_ * H_ + 255) / 256, 256, 0, stream>>>(in_proj_w, w16, 3 * H_ * H_);
  cvt_f16_kernel<<<(H_ * H_ + 255) / 256, 256, 0, stream>>>(out_proj_w, w16 + 3 * H_ * H_, H_ * H_);
  cvt_f16_kernel<<<(2 * H_ * H_ + 255) / 256, 256, 0, stream>>>(w1, w16 + 4 * H_ * H_, 2 * H_ * H_);
  cvt_f16_kernel<<<(2 * H_ * H_ + 255) / 256, 256, 0, stream>>>(w2, w16 + 6 * H_ * H_, 2 * H_ * H_);

  searchsorted_kernel<<<1, 256, 0, stream>>>(edge_batch, E_, startq);
  searchsorted_kernel<<<1, 256, 0, stream>>>(ieb, M_, startk);
  scatter_q_kernel<<<E_, H_, 0, stream>>>(edge_attr, edge_batch, startq, qc, rowidx);
  scatter_k_kernel<<<M_, H_, 0, stream>>>(edge_attr, iel, ieb, startk, kc, kmask);

  // projections
  {
    int Mq = B_ * LQ_, Mk = B_ * LK_, N = H_, Kd = H_;
    int tq = (Mq / 32) * (N / 32), tk = (Mk / 32) * (N / 32);
    gemm_wmma_kernel<false, false, false, true, false><<<(tq + 7) / 8, 256, 0, stream>>>(
        qc, nullptr, H_, w16, H_, in_proj_b, nullptr,
        Qb, 1 << 30, 0L, H_, 1, nullptr, Mq, N, Kd);
    gemm_wmma_kernel<false, false, false, true, false><<<(tk + 7) / 8, 256, 0, stream>>>(
        kc, nullptr, H_, w16 + H_ * H_, H_, in_proj_b + H_, nullptr,
        Kb, 1 << 30, 0L, H_, 1, nullptr, Mk, N, Kd);
    gemm_wmma_kernel<false, false, false, true, false><<<(tk + 7) / 8, 256, 0, stream>>>(
        kc, nullptr, H_, w16 + 2 * H_ * H_, H_, in_proj_b + 2 * H_, nullptr,
        Vt, LK_, (long)H_ * LK_, 1, LK_, nullptr, Mk, N, Kd);
  }

  // flash attention: 8192 waves -> 1024 blocks of 8 waves
  attention_kernel<<<(B_ * NH_ * (LQ_ / 16)) / 8, 256, 0, stream>>>(Qb, Kb, Vt, kmask, ctx);

  // out-proj with per-edge gather + residual
  {
    int Me = E_, N = H_, Kd = H_;
    int t = (Me / 32) * (N / 32);
    gemm_wmma_kernel<true, false, true, true, true><<<(t + 7) / 8, 256, 0, stream>>>(
        ctx, rowidx, H_, w16 + 3 * H_ * H_, H_, out_proj_b, edge_attr,
        att16, 1 << 30, 0L, H_, 1, att32, Me, N, Kd);
  }

  // FFN1: h1 = relu(att@w1^T + b1)
  {
    int Me = E_, N = 2 * H_, Kd = H_;
    int t = (Me / 32) * (N / 32);
    gemm_wmma_kernel<false, true, false, true, false><<<(t + 7) / 8, 256, 0, stream>>>(
        att16, nullptr, H_, w16 + 4 * H_ * H_, H_, b1, nullptr,
        h1b, 1 << 30, 0L, 2 * H_, 1, nullptr, Me, N, Kd);
  }

  // FFN2 + residual -> d_out
  {
    int Me = E_, N = H_, Kd = 2 * H_;
    int t = (Me / 32) * (N / 32);
    gemm_wmma_kernel<false, false, true, false, true><<<(t + 7) / 8, 256, 0, stream>>>(
        h1b, nullptr, 2 * H_, w16 + 6 * H_ * H_, 2 * H_, b2, att32,
        nullptr, 1 << 30, 0L, H_, 1, (float*)d_out, Me, N, Kd);
  }
}
